// LlamaAttention_75196287418470
// MI455X (gfx1250) — compile-verified, matrix operand order checked by reading, not run on hardware
//
#include <hip/hip_runtime.h>
#include <stdint.h>

typedef __attribute__((ext_vector_type(16))) __bf16 v16bf;
typedef __attribute__((ext_vector_type(8)))  float  v8f;

#define S_LEN  2048
#define DMODEL 4096
#define NH_Q   32
#define NH_KV  8
#define HDIM   128

// ---------------------------------------------------------------- helpers
static __device__ __forceinline__ uint16_t f2bf(float f) {
  union { float f; uint32_t u; } x; x.f = f;
  uint32_t u = x.u;
  uint32_t r = u + 0x7FFFu + ((u >> 16) & 1u);   // round-to-nearest-even
  return (uint16_t)(r >> 16);
}

// 16x32 bf16 A-fragment (or 32x16 B-fragment whose columns are contiguous
// rows of `base`) per the CDNA5 16-bit WMMA VGPR layout:
// lane L holds row (L&15); K elements (L>>4)*8..+7 and 16+(L>>4)*8..+7.
// boff/ldb are BYTE offsets kept in u32 so ISel can use saddr + v32 offset.
static __device__ __forceinline__ v16bf load_frag_g(const void* __restrict__ base,
                                                    uint32_t boff, uint32_t ldb) {
  uint32_t lane = threadIdx.x & 31;
  uint32_t o = boff + (lane & 15) * ldb + (lane >> 4) * 16;
  const char* p = (const char*)base + o;
  union { uint4 q[2]; v16bf v; } u;
  u.q[0] = *(const uint4*)p;
  u.q[1] = *(const uint4*)(p + 32);
  return u.v;
}

// Pointer version (used for LDS tiles).
static __device__ __forceinline__ v16bf load_frag_p(const __bf16* base, int ld) {
  int lane = threadIdx.x & 31;
  const __bf16* p = base + (size_t)(lane & 15) * ld + (lane >> 4) * 8;
  union { uint4 q[2]; v16bf v; } u;
  u.q[0] = *(const uint4*)p;
  u.q[1] = *(const uint4*)(p + 16);
  return u.v;
}

static __device__ __forceinline__ v8f wmma_bf16(v16bf a, v16bf b, v8f c) {
  return __builtin_amdgcn_wmma_f32_16x16x32_bf16(
      /*neg_a=*/false, a, /*neg_b=*/false, b,
      /*c_mod=*/(short)0, c, /*reuse_a=*/false, /*reuse_b=*/false);
}

// ---------------------------------------------------------------- RMSNorm
__global__ void rmsnorm_kernel(const float* __restrict__ x,
                               const float* __restrict__ w,
                               uint16_t* __restrict__ out) {
  int s = blockIdx.x;
  const float* row = x + (size_t)s * DMODEL;
  float sum = 0.f;
  for (int i = threadIdx.x; i < DMODEL; i += 256) { float v = row[i]; sum += v * v; }
  __shared__ float red[256];
  red[threadIdx.x] = sum;
  __syncthreads();
  for (int off = 128; off > 0; off >>= 1) {
    if (threadIdx.x < off) red[threadIdx.x] += red[threadIdx.x + off];
    __syncthreads();
  }
  float r = rsqrtf(red[0] * (1.0f / DMODEL) + 1e-5f);
  for (int i = threadIdx.x; i < DMODEL; i += 256)
    out[(size_t)s * DMODEL + i] = f2bf(row[i] * r * w[i]);
}

// ---------------------------------------------------------------- fp32 -> bf16
__global__ void cvt_bf16_kernel(const float* __restrict__ src,
                                uint16_t* __restrict__ dst, int n) {
  int i = blockIdx.x * 256 + threadIdx.x;
  if (i < n) dst[i] = f2bf(src[i]);
}

// ---------------------------------------------------------------- GEMM
// C[M,N] = A[M,K] * B[N,K]^T (+ addend).  A,B bf16 row-major, K contiguous.
// block = 256 threads = 8 waves; block tile 128x128; wave tile 64x32.
// Ping-pong register pipelining (K % 64 == 0): compute buffer p, reload it
// in place for k+64.  Byte offsets in u32; body kept at exactly one copy.
__global__ void __launch_bounds__(256, 1)
gemm_bf16_kernel(const uint16_t* __restrict__ Ab,
                 const uint16_t* __restrict__ Bb,
                 float* __restrict__ C,
                 const float* __restrict__ addend,
                 int M, int N, int K) {
  int wave = threadIdx.x >> 5;
  int lane = threadIdx.x & 31;
  int wm = blockIdx.y * 128 + (wave >> 2) * 64;
  int wn = blockIdx.x * 128 + (wave & 3) * 32;
  const uint32_t ldb = (uint32_t)K * 2;        // row stride in bytes

  uint32_t abase[4], bbase[2];                 // byte offsets
#pragma unroll
  for (int i = 0; i < 4; ++i) abase[i] = (uint32_t)(wm + 16 * i) * ldb;
#pragma unroll
  for (int j = 0; j < 2; ++j) bbase[j] = (uint32_t)(wn + 16 * j) * ldb;

  const v8f zf = {0.f, 0.f, 0.f, 0.f, 0.f, 0.f, 0.f, 0.f};
  v8f acc[4][2];
#pragma unroll
  for (int i = 0; i < 4; ++i)
#pragma unroll
    for (int j = 0; j < 2; ++j) acc[i][j] = zf;

  // prologue: fill both buffers (k-bytes 0 and 64)
  v16bf a[2][4], b[2][2];
#pragma unroll
  for (int i = 0; i < 4; ++i) a[0][i] = load_frag_g(Ab, abase[i], ldb);
#pragma unroll
  for (int j = 0; j < 2; ++j) b[0][j] = load_frag_g(Bb, bbase[j], ldb);
#pragma unroll
  for (int i = 0; i < 4; ++i) a[1][i] = load_frag_g(Ab, abase[i] + 64, ldb);
#pragma unroll
  for (int j = 0; j < 2; ++j) b[1][j] = load_frag_g(Bb, bbase[j] + 64, ldb);

  const uint32_t kbytes = (uint32_t)K * 2;     // 32 elems = 64 bytes per step
  uint32_t kb = 0;
#pragma unroll 1
  for (; kb + 128 < kbytes; kb += 128) {
    __builtin_prefetch((const char*)Ab + abase[0] + kb + 2048, 0, 0);
    __builtin_prefetch((const char*)Bb + bbase[0] + kb + 2048, 0, 0);
    // compute buffer 0 (kb), refill it for kb+128
#pragma unroll
    for (int i = 0; i < 4; ++i) {
      acc[i][0] = wmma_bf16(a[0][i], b[0][0], acc[i][0]);
      acc[i][1] = wmma_bf16(a[0][i], b[0][1], acc[i][1]);
    }
#pragma unroll
    for (int i = 0; i < 4; ++i) a[0][i] = load_frag_g(Ab, abase[i] + kb + 128, ldb);
#pragma unroll
    for (int j = 0; j < 2; ++j) b[0][j] = load_frag_g(Bb, bbase[j] + kb + 128, ldb);
    // compute buffer 1 (kb+64), refill it for kb+192
#pragma unroll
    for (int i = 0; i < 4; ++i) {
      acc[i][0] = wmma_bf16(a[1][i], b[1][0], acc[i][0]);
      acc[i][1] = wmma_bf16(a[1][i], b[1][1], acc[i][1]);
    }
#pragma unroll
    for (int i = 0; i < 4; ++i) a[1][i] = load_frag_g(Ab, abase[i] + kb + 192, ldb);
#pragma unroll
    for (int j = 0; j < 2; ++j) b[1][j] = load_frag_g(Bb, bbase[j] + kb + 192, ldb);
  }
  // epilogue: last two k-steps, no further loads
#pragma unroll
  for (int i = 0; i < 4; ++i) {
    acc[i][0] = wmma_bf16(a[0][i], b[0][0], acc[i][0]);
    acc[i][1] = wmma_bf16(a[0][i], b[0][1], acc[i][1]);
  }
#pragma unroll
  for (int i = 0; i < 4; ++i) {
    acc[i][0] = wmma_bf16(a[1][i], b[1][0], acc[i][0]);
    acc[i][1] = wmma_bf16(a[1][i], b[1][1], acc[i][1]);
  }

  uint32_t hf = (lane >> 4), cl = (lane & 15);
#pragma unroll
  for (int i = 0; i < 4; ++i)
#pragma unroll
    for (int j = 0; j < 2; ++j)
#pragma unroll
      for (int r = 0; r < 8; ++r) {
        uint32_t row = wm + 16 * i + r + 8 * hf;
        uint32_t cn  = wn + 16 * j + cl;
        uint32_t ob  = (row * (uint32_t)N + cn) * 4;   // byte offset
        float v = acc[i][j][r];
        if (addend) v += *(const float*)((const char*)addend + ob);
        *(float*)((char*)C + ob) = v;
      }
}

// ---------------------------------------------------------------- RoPE
// src fp32 [S, NH*HD]; dst bf16 [NH][S][HD]; interleaved rotate-half.
__global__ void rope_kernel(const float* __restrict__ src,
                            const float* __restrict__ cosb,
                            const float* __restrict__ sinb,
                            uint16_t* __restrict__ dst, int NH) {
  int idx = blockIdx.x * 256 + threadIdx.x;
  int total = S_LEN * NH * (HDIM / 2);
  if (idx >= total) return;
  int i  = idx & 63;            // pair index (HD/2 = 64)
  int t  = idx >> 6;
  int nh = t % NH;
  int s  = t / NH;
  int d0 = 2 * i;
  float c  = cosb[s * HDIM + d0];
  float sn = sinb[s * HDIM + d0];
  uint32_t sb = (uint32_t)s * (NH * HDIM) + nh * HDIM;
  float x0 = src[sb + d0];
  float x1 = src[sb + d0 + 1];
  uint32_t db = ((uint32_t)nh * S_LEN + s) * HDIM;
  dst[db + d0]     = f2bf(x0 * c - x1 * sn);
  dst[db + d0 + 1] = f2bf(x1 * c + x0 * sn);
}

// ---------------------------------------------------------------- V transpose
// v fp32 [S, HKV*HD] -> vT bf16 [HKV][HD][S]
__global__ void vtrans_kernel(const float* __restrict__ v,
                              uint16_t* __restrict__ dst) {
  int idx = blockIdx.x * 256 + threadIdx.x;
  if (idx >= S_LEN * NH_KV * HDIM) return;
  int d  = idx & (HDIM - 1);
  int t  = idx >> 7;
  int hk = t % NH_KV;
  int s  = t / NH_KV;
  dst[((uint32_t)hk * HDIM + d) * S_LEN + s] =
      f2bf(v[(uint32_t)s * (NH_KV * HDIM) + hk * HDIM + d]);
}

// ---------------------------------------------------------------- Flash attention
// One wave per (head, 16-row q tile). Q bf16 [H][S][HD], K bf16 [HKV][S][HD],
// V^T bf16 [HKV][HD][S].  Output o bf16 [S][H*HD].
__global__ void __launch_bounds__(256, 1)
attn_kernel(const uint16_t* __restrict__ Qb,
            const uint16_t* __restrict__ Kb,
            const uint16_t* __restrict__ Vt,
            uint16_t* __restrict__ Ob) {
  __shared__ __align__(16) uint16_t ldsp[8][16 * 32];

  int wave = threadIdx.x >> 5;
  int lane = threadIdx.x & 31;
  int task = blockIdx.x * 8 + wave;
  int qt   = task & 127;        // 2048/16 q tiles
  int h    = task >> 7;         // q head
  int hkv  = h >> 2;            // G = 4
  int q0   = qt * 16;
  int hf = lane >> 4, cl = lane & 15;
  const float scale = 0.08838834764831845f;   // 1/sqrt(128)

  // byte offsets (u32, saddr-friendly)
  uint32_t qoff = ((uint32_t)h * S_LEN + q0) * HDIM * 2;
  uint32_t koff = (uint32_t)hkv * S_LEN * HDIM * 2;
  uint32_t voff = (uint32_t)hkv * HDIM * S_LEN * 2;

  v16bf qf[4];
#pragma unroll
  for (int d = 0; d < 4; ++d) qf[d] = load_frag_g(Qb, qoff + d * 64, HDIM * 2);

  const v8f zf = {0.f, 0.f, 0.f, 0.f, 0.f, 0.f, 0.f, 0.f};
  v8f o[8];
#pragma unroll
  for (int d = 0; d < 8; ++d) o[d] = zf;
  float m[8], l[8];
#pragma unroll
  for (int r = 0; r < 8; ++r) { m[r] = -3.0e38f; l[r] = 0.f; }

  int kend = q0 + 16;                    // causal: need k <= q0+15
#pragma unroll 1
  for (int k0 = 0; k0 < kend; k0 += 32) {
    // ---- scores: two 16-wide tiles, K-depth 128 via 4 WMMA each
    v8f s0 = zf, s1 = zf;
#pragma unroll
    for (int d = 0; d < 4; ++d) {
      v16bf kb = load_frag_g(Kb, koff + (uint32_t)k0 * (HDIM * 2) + d * 64, HDIM * 2);
      s0 = wmma_bf16(qf[d], kb, s0);
    }
    if (k0 < q0) {                       // second tile has any live columns
#pragma unroll
      for (int d = 0; d < 4; ++d) {
        v16bf kb = load_frag_g(Kb, koff + (uint32_t)(k0 + 16) * (HDIM * 2) + d * 64, HDIM * 2);
        s1 = wmma_bf16(qf[d], kb, s1);
      }
    }

    // ---- scale + causal mask + row max
    float tmax[8];
#pragma unroll
    for (int r = 0; r < 8; ++r) {
      int qrow = q0 + r + 8 * hf;
      float a0 = (qrow >= k0 + cl)      ? s0[r] * scale : -1.0e30f;
      float a1 = (qrow >= k0 + 16 + cl) ? s1[r] * scale : -1.0e30f;
      s0[r] = a0; s1[r] = a1;
      tmax[r] = fmaxf(a0, a1);
    }
#pragma unroll
    for (int mk = 1; mk < 16; mk <<= 1)
#pragma unroll
      for (int r = 0; r < 8; ++r)
        tmax[r] = fmaxf(tmax[r], __shfl_xor(tmax[r], mk, 32));

    // ---- online softmax update
    float alpha[8], rsum[8];
#pragma unroll
    for (int r = 0; r < 8; ++r) {
      float mn = fmaxf(m[r], tmax[r]);
      alpha[r] = __expf(m[r] - mn);
      m[r] = mn;
      float p0 = __expf(s0[r] - mn);
      float p1 = __expf(s1[r] - mn);
      s0[r] = p0; s1[r] = p1;
      rsum[r] = p0 + p1;
    }
#pragma unroll
    for (int mk = 1; mk < 16; mk <<= 1)
#pragma unroll
      for (int r = 0; r < 8; ++r)
        rsum[r] += __shfl_xor(rsum[r], mk, 32);
#pragma unroll
    for (int r = 0; r < 8; ++r) l[r] = l[r] * alpha[r] + rsum[r];
#pragma unroll
    for (int d = 0; d < 8; ++d)
#pragma unroll
      for (int r = 0; r < 8; ++r) o[d][r] *= alpha[r];

    // ---- P: C-frag layout -> 16x32 row-major bf16 in LDS -> A-frag
    uint16_t* lp = &ldsp[wave][0];
#pragma unroll
    for (int r = 0; r < 8; ++r) {
      int row = r + 8 * hf;
      lp[row * 32 + cl]      = f2bf(s0[r]);
      lp[row * 32 + 16 + cl] = f2bf(s1[r]);
    }
    asm volatile("" ::: "memory");   // keep ds_store before ds_load (HW keeps LDS in-order per wave)
    v16bf pf = load_frag_p((const __bf16*)lp, 32);

    // ---- O += P * V : 8 WMMA over HD
#pragma unroll
    for (int dt = 0; dt < 8; ++dt) {
      v16bf vb = load_frag_g(Vt, voff + (uint32_t)(dt * 16) * (S_LEN * 2) + k0 * 2, S_LEN * 2);
      o[dt] = wmma_bf16(pf, vb, o[dt]);
    }
  }

  // ---- normalize + store bf16 [S][H*HD]
#pragma unroll
  for (int dt = 0; dt < 8; ++dt)
#pragma unroll
    for (int r = 0; r < 8; ++r) {
      uint32_t row = q0 + r + 8 * hf;
      uint32_t d   = dt * 16 + cl;
      Ob[row * (NH_Q * HDIM) + h * HDIM + d] = f2bf(o[dt][r] / l[r]);
    }
}

// ---------------------------------------------------------------- launch
extern "C" void kernel_launch(void* const* d_in, const int* in_sizes, int n_in,
                              void* d_out, int out_size, void* d_ws, size_t ws_size,
                              hipStream_t stream) {
  const float* hidden = (const float*)d_in[0];
  const float* cosb   = (const float*)d_in[1];
  const float* sinb   = (const float*)d_in[2];
  const float* rmsw   = (const float*)d_in[3];
  const float* wq     = (const float*)d_in[4];
  const float* wk     = (const float*)d_in[5];
  const float* wv     = (const float*)d_in[6];
  const float* wo     = (const float*)d_in[7];
  // d_in[8] k_cache, d_in[9] v_cache, d_in[10] position_ids: identity mapping here.
  float* out = (float*)d_out;

  char* ws = (char*)d_ws;
  size_t off = 0;
  auto alloc = [&](size_t bytes) -> char* {
    char* p = ws + off;
    off += (bytes + 255) & ~(size_t)255;
    return p;
  };
  uint16_t* hsb = (uint16_t*)alloc((size_t)S_LEN * DMODEL * 2);
  uint16_t* wqb = (uint16_t*)alloc((size_t)DMODEL * DMODEL * 2);
  uint16_t* wkb = (uint16_t*)alloc((size_t)NH_KV * HDIM * DMODEL * 2);
  uint16_t* wvb = (uint16_t*)alloc((size_t)NH_KV * HDIM * DMODEL * 2);
  uint16_t* wob = (uint16_t*)alloc((size_t)DMODEL * DMODEL * 2);
  float*    qf  = (float*)   alloc((size_t)S_LEN * NH_Q * HDIM * 4);
  float*    kf  = (float*)   alloc((size_t)S_LEN * NH_KV * HDIM * 4);
  float*    vf  = (float*)   alloc((size_t)S_LEN * NH_KV * HDIM * 4);
  uint16_t* qrb = (uint16_t*)alloc((size_t)NH_Q * S_LEN * HDIM * 2);
  uint16_t* krb = (uint16_t*)alloc((size_t)NH_KV * S_LEN * HDIM * 2);
  uint16_t* vtb = (uint16_t*)alloc((size_t)NH_KV * HDIM * S_LEN * 2);
  uint16_t* ob  = (uint16_t*)alloc((size_t)S_LEN * NH_Q * HDIM * 2);

  // 1. RMSNorm -> bf16
  rmsnorm_kernel<<<S_LEN, 256, 0, stream>>>(hidden, rmsw, hsb);

  // 2. weights -> bf16
  int nqo = DMODEL * DMODEL, nkv = NH_KV * HDIM * DMODEL;
  cvt_bf16_kernel<<<(nqo + 255) / 256, 256, 0, stream>>>(wq, wqb, nqo);
  cvt_bf16_kernel<<<(nkv + 255) / 256, 256, 0, stream>>>(wk, wkb, nkv);
  cvt_bf16_kernel<<<(nkv + 255) / 256, 256, 0, stream>>>(wv, wvb, nkv);
  cvt_bf16_kernel<<<(nqo + 255) / 256, 256, 0, stream>>>(wo, wob, nqo);

  // 3. QKV projections
  gemm_bf16_kernel<<<dim3(DMODEL / 128, S_LEN / 128), 256, 0, stream>>>(
      hsb, wqb, qf, nullptr, S_LEN, DMODEL, DMODEL);
  gemm_bf16_kernel<<<dim3((NH_KV * HDIM) / 128, S_LEN / 128), 256, 0, stream>>>(
      hsb, wkb, kf, nullptr, S_LEN, NH_KV * HDIM, DMODEL);
  gemm_bf16_kernel<<<dim3((NH_KV * HDIM) / 128, S_LEN / 128), 256, 0, stream>>>(
      hsb, wvb, vf, nullptr, S_LEN, NH_KV * HDIM, DMODEL);

  // 4. RoPE + layout transforms
  int nq = S_LEN * NH_Q * (HDIM / 2), nk = S_LEN * NH_KV * (HDIM / 2);
  rope_kernel<<<(nq + 255) / 256, 256, 0, stream>>>(qf, cosb, sinb, qrb, NH_Q);
  rope_kernel<<<(nk + 255) / 256, 256, 0, stream>>>(kf, cosb, sinb, krb, NH_KV);
  int nv = S_LEN * NH_KV * HDIM;
  vtrans_kernel<<<(nv + 255) / 256, 256, 0, stream>>>(vf, vtb);

  // 5. flash attention: 128 q-tiles * 32 heads / 8 waves per block
  attn_kernel<<<(128 * NH_Q) / 8, 256, 0, stream>>>(qrb, krb, vtb, ob);

  // 6. output projection + residual -> d_out
  gemm_bf16_kernel<<<dim3(DMODEL / 128, S_LEN / 128), 256, 0, stream>>>(
      ob, wob, out, hidden, S_LEN, DMODEL, DMODEL);
}
